// Attn_49881750176227
// MI455X (gfx1250) — matrix operand order repaired and checked by reading, not compiled, not run on hardware
//
#include <hip/hip_runtime.h>
#include <hip/hip_bf16.h>

// ---------------------------------------------------------------------------
// Causal self-attention block for MI455X (gfx1250, wave32, WMMA).
// - All GEMMs via V_WMMA_F32_16X16X32_BF16 (bf16 in, fp32 accumulate)
// - native v_cvt_pk_bf16_f32 conversions, b128 global loads
// - 32x64 register blocking in the projection GEMMs
// - flash-attention streaming softmax; K / V^T tiles double-buffered into LDS
//   with GLOBAL_LOAD_ASYNC_TO_LDS_B128 (ASYNCcnt)
// ---------------------------------------------------------------------------

typedef __attribute__((ext_vector_type(16))) __bf16 v16bf;
typedef __attribute__((ext_vector_type(8)))  float  v8f;
typedef __attribute__((ext_vector_type(2)))  float  f32x2;
typedef __attribute__((ext_vector_type(2)))  __bf16 bf16x2;
typedef int v4i __attribute__((vector_size(16)));

union BfFrag {
    v16bf        v;
    unsigned int u32[8];
};

__device__ __forceinline__ unsigned int cvt_pk_bf16(f32x2 v) {
    bf16x2 b = __builtin_convertvector(v, bf16x2);   // v_cvt_pk_bf16_f32
    unsigned int u; __builtin_memcpy(&u, &b, 4);
    return u;
}
__device__ __forceinline__ unsigned short bf16_bits(float f) {
    __bf16 h = (__bf16)f;
    unsigned short u; __builtin_memcpy(&u, &h, 2);
    return u;
}
__device__ __forceinline__ f32x2 ld2(const float* p) {      // global_load_b64
    return *(const f32x2*)p;
}

// ISA 7.12.2 16-bit A-matrix 16x32: VGPR j, lane-half hi -> K offset
__device__ __forceinline__ int a_koff(int j, int hi) {
    return ((j >> 2) << 4) + (hi << 3) + ((j & 3) << 1);
}
// 16-bit B-matrix 32x16: same K order as A (VGPR j, lane-half hi -> K offset)
__device__ __forceinline__ int b_koff(int j, int hi) {
    return ((j >> 2) << 4) + (hi << 3) + ((j & 3) << 1);
}

#define NB     4
#define NSEQ   2048
#define DMODEL 1024
#define NH     16
#define DH     64

// ---- CDNA5 async global->LDS copy (ASYNCcnt), with portable fallback ------
#if __has_builtin(__builtin_amdgcn_global_load_async_to_lds_b128)
#define HAVE_ASYNC_LDS 1
#define ASYNC_B128(srcp, dstp)                                                 \
    __builtin_amdgcn_global_load_async_to_lds_b128(                            \
        (__attribute__((address_space(1))) v4i*)(srcp),                        \
        (__attribute__((address_space(3))) v4i*)(dstp), 0, 0)
#define WAIT_ASYNC(n) asm volatile("s_wait_asynccnt %0" ::"i"(n) : "memory")
#else
#define HAVE_ASYNC_LDS 0
#define ASYNC_B128(srcp, dstp)                                                 \
    do { *(uint4*)(dstp) = *(const uint4*)(srcp); } while (0)
#define WAIT_ASYNC(n) asm volatile("" ::: "memory")
#endif
#define WAIT_DS0() asm volatile("s_wait_dscnt 0" ::: "memory")

// ---------------------------------------------------------------------------
// Kernel 1: qkv = X @ W1^T + b1. One wave -> 32x64 tile (8 WMMA / K-step).
// Scatter: Q -> bf16 ws (B,H,N,DH) pre-scaled by 1/sqrt(DH)
//          K -> fp32 out (B,H,DH,N)  +  bf16 ws (B,H,N,DH)   [for Q.K^T]
//          V -> fp32 out (B,H,N,DH)  +  bf16 ws (B,H,DH,N)   [for P.V]
// ---------------------------------------------------------------------------
__global__ __launch_bounds__(256) void qkv_gemm_kernel(
    const float* __restrict__ X, const float* __restrict__ W,
    const float* __restrict__ bias,
    unsigned short* __restrict__ qb, unsigned short* __restrict__ kb,
    unsigned short* __restrict__ vtb,
    float* __restrict__ k_out, float* __restrict__ v_out)
{
    const int lane = threadIdx.x & 31;
    const int wave = __builtin_amdgcn_readfirstlane(
        blockIdx.x * (blockDim.x >> 5) + (threadIdx.x >> 5));
    const int NT = (3 * DMODEL) / 64;            // 48 tile-cols
    const int mt = wave / NT, nt = wave % NT;
    const int m0 = mt << 5, n0 = nt << 6;
    const int r = lane & 15, hi = lane >> 4;

    v8f acc[2][4] = {};
    const float* xr0 = X + (size_t)(m0 + r) * DMODEL;
    const float* xr1 = xr0 + (size_t)16 * DMODEL;
    const float* wr = W + (size_t)(n0 + r) * DMODEL;

    for (int k0 = 0; k0 < DMODEL; k0 += 32) {
        __builtin_prefetch(xr0 + k0 + 32, 0, 1);
        __builtin_prefetch(xr1 + k0 + 32, 0, 1);
        __builtin_prefetch(wr + k0 + 32, 0, 1);
        BfFrag a[2], b[4];
#pragma unroll
        for (int j = 0; j < 8; ++j) {
            const int ka = k0 + a_koff(j, hi);
            a[0].u32[j] = cvt_pk_bf16(ld2(xr0 + ka));
            a[1].u32[j] = cvt_pk_bf16(ld2(xr1 + ka));
            const int kbo = k0 + b_koff(j, hi);
#pragma unroll
            for (int ni = 0; ni < 4; ++ni)
                b[ni].u32[j] = cvt_pk_bf16(ld2(wr + (size_t)ni * 16 * DMODEL + kbo));
        }
#pragma unroll
        for (int mi = 0; mi < 2; ++mi)
#pragma unroll
            for (int ni = 0; ni < 4; ++ni)
                acc[mi][ni] = __builtin_amdgcn_wmma_f32_16x16x32_bf16(
                    false, a[mi].v, false, b[ni].v, (short)0, acc[mi][ni],
                    false, false);
    }

    // ---- scatter (segment is wave-uniform: n0 is 64-aligned, segs at 1024) --
#pragma unroll
    for (int ni = 0; ni < 4; ++ni) {
        const int n = n0 + ni * 16 + r;          // qkv feature index
        const float bn = bias[n];
#pragma unroll
        for (int mi = 0; mi < 2; ++mi) {
#pragma unroll
            for (int i = 0; i < 8; ++i) {
                const int m   = m0 + mi * 16 + (hi << 3) + i;
                const int bi  = m >> 11;
                const int pos = m & 2047;
                const float val = acc[mi][ni][i] + bn;
                if (n0 < DMODEL) {               // Q
                    const int h = n >> 6, dh = n & 63;
                    const size_t bh = (size_t)bi * NH + h;
                    qb[(bh * NSEQ + pos) * DH + dh] = bf16_bits(val * 0.125f);
                } else if (n0 < 2 * DMODEL) {    // K
                    const int f = n - DMODEL, h = f >> 6, d = f & 63;
                    const size_t bh = (size_t)bi * NH + h;
                    k_out[(bh * DH + d) * NSEQ + pos] = val;
                    kb[(bh * NSEQ + pos) * DH + d]    = bf16_bits(val);
                } else {                         // V
                    const int f = n - 2 * DMODEL, h = f >> 6, d = f & 63;
                    const size_t bh = (size_t)bi * NH + h;
                    v_out[(bh * NSEQ + pos) * DH + d] = val;
                    vtb[(bh * DH + d) * NSEQ + pos]   = bf16_bits(val);
                }
            }
        }
    }
}

// ---------------------------------------------------------------------------
// Per-wave async staging of one 32-position K tile (32 rows x 64 bf16) and
// V^T tile (64 rows x 32 bf16) into LDS: 16 x B128 async ops.
// ---------------------------------------------------------------------------
__device__ __forceinline__ void stage_kv(
    const unsigned short* __restrict__ kbase,
    const unsigned short* __restrict__ vtbase, int kt,
    unsigned short* kdst /*[32][64]*/, unsigned short* vdst /*[64][32]*/,
    int lane)
{
    // K: lane copies row (kt+lane), 128 bytes = 8 x b128
    const unsigned short* ksrc = kbase + (size_t)(kt + lane) * DH;
    unsigned short* kd = kdst + lane * DH;
#pragma unroll
    for (int c = 0; c < 8; ++c)
        ASYNC_B128(ksrc + c * 8, kd + c * 8);
    // V^T: lane copies rows 2*lane, 2*lane+1, 64 bytes each = 8 x b128
    const unsigned short* vs0 = vtbase + (size_t)(2 * lane) * NSEQ + kt;
    unsigned short* vd = vdst + (2 * lane) * 32;
#pragma unroll
    for (int c = 0; c < 4; ++c) {
        ASYNC_B128(vs0 + c * 8, vd + c * 8);
        ASYNC_B128(vs0 + NSEQ + c * 8, vd + 32 + c * 8);
    }
}

// ---------------------------------------------------------------------------
// Kernel 2: flash attention. One wave per (b,h,16-row tile); streams 32
// key/value positions per step with online softmax. K/V^T tiles are
// double-buffered in LDS via async global->LDS B128 copies.
// ---------------------------------------------------------------------------
__global__ __launch_bounds__(128) void attn_kernel(
    const unsigned short* __restrict__ qb, const unsigned short* __restrict__ kb,
    const unsigned short* __restrict__ vtb, unsigned short* __restrict__ ob)
{
    __shared__ unsigned short kstage[4][2][32 * DH];   // 4 waves x 2 bufs x 4KB
    __shared__ unsigned short vstage[4][2][DH * 32];   // 4 waves x 2 bufs x 4KB
    __shared__ unsigned short pbuf[4][16][34];         // P transpose bounce

    const int lane  = threadIdx.x & 31;
    const int wslot = threadIdx.x >> 5;
    const int wave  = __builtin_amdgcn_readfirstlane(blockIdx.x * 4 + wslot);
    const int rt = wave & 127;                    // row tile
    const int bh = wave >> 7;                     // b*16+h, 0..63
    const int r0 = rt << 4;
    const int r = lane & 15, hi = lane >> 4;

    const unsigned short* qbase  = qb  + (size_t)bh * NSEQ * DH;
    const unsigned short* kbase  = kb  + (size_t)bh * NSEQ * DH;
    const unsigned short* vtbase = vtb + (size_t)bh * DH * NSEQ;

    unsigned short* kst[2] = { &kstage[wslot][0][0], &kstage[wslot][1][0] };
    unsigned short* vst[2] = { &vstage[wslot][0][0], &vstage[wslot][1][0] };

    // Q A-fragments (16x64 -> two K=32 fragments), pre-scaled by 1/sqrt(DH)
    BfFrag qf[2];
#pragma unroll
    for (int t = 0; t < 2; ++t) {
        const unsigned short* qr = qbase + (size_t)(r0 + r) * DH + t * 32;
#pragma unroll
        for (int j = 0; j < 8; ++j)
            qf[t].u32[j] = *(const unsigned int*)(qr + a_koff(j, hi));
    }

    v8f o[4] = {};
    float mrow[8], lrow[8];
#pragma unroll
    for (int i = 0; i < 8; ++i) { mrow[i] = -1e30f; lrow[i] = 0.0f; }

    const int nsteps = (r0 + 16 + 31) >> 5;       // causal: cols <= r0+15

    stage_kv(kbase, vtbase, 0, kst[0], vst[0], lane);   // prologue: buffer 0

    for (int s = 0; s < nsteps; ++s) {
        const int kt = s << 5;
        const int cur = s & 1;

        // stage next tile into the other buffer; gate current buffer.
        if (s + 1 < nsteps) {
            WAIT_DS0();   // prior DS reads of that buffer must retire first
            stage_kv(kbase, vtbase, kt + 32, kst[cur ^ 1], vst[cur ^ 1], lane);
            WAIT_ASYNC(16);   // 16 in flight for next buf => current buf done
        } else {
            WAIT_ASYNC(0);
        }
        const unsigned short* kcur = kst[cur];
        const unsigned short* vcur = vst[cur];

        // ---- S = Q @ K^T for two 16-column tiles (K-depth = DH = 64) ----
        v8f sfr[2];
#pragma unroll
        for (int t = 0; t < 2; ++t) {
            const unsigned short* krow = kcur + (t * 16 + r) * DH;
            BfFrag kf0, kf1;
#pragma unroll
            for (int j = 0; j < 8; ++j) {
                const int d = b_koff(j, hi);
                kf0.u32[j] = *(const unsigned int*)(krow + d);
                kf1.u32[j] = *(const unsigned int*)(krow + 32 + d);
            }
            v8f sacc = {};
            sacc = __builtin_amdgcn_wmma_f32_16x16x32_bf16(
                false, qf[0].v, false, kf0.v, (short)0, sacc, false, false);
            sacc = __builtin_amdgcn_wmma_f32_16x16x32_bf16(
                false, qf[1].v, false, kf1.v, (short)0, sacc, false, false);
            sfr[t] = sacc;
        }

        // ---- causal mask + online softmax ----
        float scl[8];
#pragma unroll
        for (int i = 0; i < 8; ++i) {
            const int row = r0 + (hi << 3) + i;
            const int c0 = kt + r;
            sfr[0][i] = (c0 > row)      ? -1e30f : sfr[0][i];
            sfr[1][i] = (c0 + 16 > row) ? -1e30f : sfr[1][i];

            float t = fmaxf(sfr[0][i], sfr[1][i]);
#pragma unroll
            for (int mk = 1; mk < 16; mk <<= 1) t = fmaxf(t, __shfl_xor(t, mk, 32));
            const float mnew = fmaxf(mrow[i], t);
            const float p0 = __expf(sfr[0][i] - mnew);
            const float p1 = __expf(sfr[1][i] - mnew);
            sfr[0][i] = p0; sfr[1][i] = p1;
            float rs = p0 + p1;
#pragma unroll
            for (int mk = 1; mk < 16; mk <<= 1) rs += __shfl_xor(rs, mk, 32);
            const float sc = __expf(mrow[i] - mnew);
            lrow[i] = lrow[i] * sc + rs;
            mrow[i] = mnew;
            scl[i] = sc;
        }
#pragma unroll
        for (int g = 0; g < 4; ++g)
#pragma unroll
            for (int i = 0; i < 8; ++i) o[g][i] *= scl[i];

        // ---- P: C-layout -> A-layout via per-wave LDS bounce (bf16) ----
#pragma unroll
        for (int i = 0; i < 8; ++i) {
            const int row = (hi << 3) + i;
            pbuf[wslot][row][r]      = bf16_bits(sfr[0][i]);
            pbuf[wslot][row][16 + r] = bf16_bits(sfr[1][i]);
        }
        WAIT_DS0();
        BfFrag pa;
#pragma unroll
        for (int j = 0; j < 8; ++j)
            pa.u32[j] = *(const unsigned int*)&pbuf[wslot][r][a_koff(j, hi)];

        // ---- O += P(16x32) @ V(32x64): B-frags from V^T LDS tile ----
#pragma unroll
        for (int g = 0; g < 4; ++g) {
            const unsigned short* vrow = vcur + (g * 16 + r) * 32;
            BfFrag vf;
#pragma unroll
            for (int j = 0; j < 8; ++j)
                vf.u32[j] = *(const unsigned int*)(vrow + b_koff(j, hi));
            o[g] = __builtin_amdgcn_wmma_f32_16x16x32_bf16(
                false, pa.v, false, vf.v, (short)0, o[g], false, false);
        }
    }

    // ---- normalize, store out1 as bf16 in (B, N, H*DH) ----
    const int b_ = bh >> 4, h_ = bh & 15;
#pragma unroll
    for (int i = 0; i < 8; ++i) {
        const float inv = 1.0f / lrow[i];
        const int row = r0 + (hi << 3) + i;
        unsigned short* orow = ob + ((size_t)(b_ * NSEQ + row) * DMODEL) + h_ * DH;
#pragma unroll
        for (int g = 0; g < 4; ++g)
            orow[g * 16 + r] = bf16_bits(o[g][i] * inv);
    }
}

// ---------------------------------------------------------------------------
// Kernel 3: y = out1 @ out_w^T + out_b. One wave -> 32x64 tile.
// ---------------------------------------------------------------------------
__global__ __launch_bounds__(256) void out_gemm_kernel(
    const unsigned short* __restrict__ A, const float* __restrict__ W,
    const float* __restrict__ bias, float* __restrict__ Y)
{
    const int lane = threadIdx.x & 31;
    const int wave = __builtin_amdgcn_readfirstlane(
        blockIdx.x * (blockDim.x >> 5) + (threadIdx.x >> 5));
    const int NT = DMODEL / 64;                   // 16
    const int mt = wave / NT, nt = wave % NT;
    const int m0 = mt << 5, n0 = nt << 6;
    const int r = lane & 15, hi = lane >> 4;

    v8f acc[2][4] = {};
    const unsigned short* ar0 = A + (size_t)(m0 + r) * DMODEL;
    const unsigned short* ar1 = ar0 + (size_t)16 * DMODEL;
    const float* wr = W + (size_t)(n0 + r) * DMODEL;

    for (int k0 = 0; k0 < DMODEL; k0 += 32) {
        __builtin_prefetch(wr + k0 + 32, 0, 1);
        BfFrag a[2], b[4];
#pragma unroll
        for (int j = 0; j < 8; ++j) {
            const int ka = k0 + a_koff(j, hi);
            a[0].u32[j] = *(const unsigned int*)(ar0 + ka);
            a[1].u32[j] = *(const unsigned int*)(ar1 + ka);
            const int kbo = k0 + b_koff(j, hi);
#pragma unroll
            for (int ni = 0; ni < 4; ++ni)
                b[ni].u32[j] = cvt_pk_bf16(ld2(wr + (size_t)ni * 16 * DMODEL + kbo));
        }
#pragma unroll
        for (int mi = 0; mi < 2; ++mi)
#pragma unroll
            for (int ni = 0; ni < 4; ++ni)
                acc[mi][ni] = __builtin_amdgcn_wmma_f32_16x16x32_bf16(
                    false, a[mi].v, false, b[ni].v, (short)0, acc[mi][ni],
                    false, false);
    }

#pragma unroll
    for (int ni = 0; ni < 4; ++ni) {
        const int n = n0 + ni * 16 + r;
        const float bn = bias[n];
#pragma unroll
        for (int mi = 0; mi < 2; ++mi)
#pragma unroll
            for (int i = 0; i < 8; ++i) {
                const int m = m0 + mi * 16 + (hi << 3) + i;
                Y[(size_t)m * DMODEL + n] = acc[mi][ni][i] + bn;
            }
    }
}

// ---------------------------------------------------------------------------
extern "C" void kernel_launch(void* const* d_in, const int* in_sizes, int n_in,
                              void* d_out, int out_size, void* d_ws, size_t ws_size,
                              hipStream_t stream) {
    (void)in_sizes; (void)n_in; (void)out_size; (void)ws_size;
    const float* X     = (const float*)d_in[0];
    // d_in[1] = past_k (empty), d_in[2] = past_v (empty)
    const float* W1_w  = (const float*)d_in[3];
    const float* W1_b  = (const float*)d_in[4];
    const float* out_w = (const float*)d_in[5];
    const float* out_b = (const float*)d_in[6];

    float* y     = (float*)d_out;
    float* k_out = y + (size_t)NB * NSEQ * DMODEL;
    float* v_out = k_out + (size_t)NB * NH * DH * NSEQ;

    const size_t SZ = (size_t)NB * NSEQ * DMODEL;     // 8388608 elements
    unsigned short* qb  = (unsigned short*)d_ws;      // bf16 Q   (B,H,N,DH)
    unsigned short* kb  = qb  + SZ;                   // bf16 K   (B,H,N,DH)
    unsigned short* vtb = kb  + SZ;                   // bf16 V^T (B,H,DH,N)
    unsigned short* ob  = vtb + SZ;                   // bf16 out1 (B,N,D)

    // 8192x3072 in 32x64 tiles -> 12288 waves -> 1536 blocks of 8 waves
    qkv_gemm_kernel<<<1536, 256, 0, stream>>>(X, W1_w, W1_b, qb, kb, vtb,
                                              k_out, v_out);
    // 4*16*128 = 8192 waves -> 2048 blocks of 4 waves
    attn_kernel<<<2048, 128, 0, stream>>>(qb, kb, vtb, ob);
    // 8192x1024 in 32x64 tiles -> 4096 waves -> 512 blocks of 8 waves
    out_gemm_kernel<<<512, 256, 0, stream>>>(ob, out_w, out_b, y);
}
